// AgentLearningDecoder_10342281248816
// MI455X (gfx1250) — matrix operand order repaired and checked by reading, not compiled, not run on hardware
//
#include <hip/hip_runtime.h>
#include <hip/hip_bf16.h>

typedef __attribute__((ext_vector_type(16))) __bf16 v16bf;
typedef __attribute__((ext_vector_type(8)))  __bf16 v8bf;
typedef __attribute__((ext_vector_type(8)))  float  v8f;

#define DK   256
#define HWN  4096
#define TN   256
#define BN   32

__device__ __forceinline__ v8f zero8() {
  v8f z;
#pragma unroll
  for (int i = 0; i < 8; ++i) z[i] = 0.0f;
  return z;
}

// ---- fragment loaders -------------------------------------------------------
// A-matrix 16x32 bf16 (ISA 7.12.2): lanes 0-15 hold K={0..7,16..23},
// lanes 16-31 hold K={8..15,24..31}; M = lane%16.
__device__ __forceinline__ v16bf loadA_f32(const float* __restrict__ row, int kc, int h) {
  const float* p0 = row + kc * 32 + h * 8;
  const float* p1 = row + kc * 32 + 16 + h * 8;
  v16bf r;
#pragma unroll
  for (int i = 0; i < 8; ++i) { r[i] = (__bf16)p0[i]; r[8 + i] = (__bf16)p1[i]; }
  return r;
}
__device__ __forceinline__ v16bf loadA_bf(const __bf16* __restrict__ row, int kc, int h) {
  v8bf a = *(const v8bf*)(row + kc * 32 + h * 8);
  v8bf b = *(const v8bf*)(row + kc * 32 + 16 + h * 8);
  v16bf r;
#pragma unroll
  for (int i = 0; i < 8; ++i) { r[i] = a[i]; r[8 + i] = b[i]; }
  return r;
}
// B-matrix 32x16 bf16: lanes 0-15 hold K={0..7,16..23}, lanes 16-31 hold K={8..15,24..31};
// N (column) = lane%16.
__device__ __forceinline__ v16bf loadB_bf(const __bf16* row, int kc, int h) {
  v8bf a = *(const v8bf*)(row + kc * 32 + h * 8);
  v8bf b = *(const v8bf*)(row + kc * 32 + 16 + h * 8);
  v16bf r;
#pragma unroll
  for (int i = 0; i < 8; ++i) { r[i] = a[i]; r[8 + i] = b[i]; }
  return r;
}

// ---- async copy helpers (ASYNCcnt-tracked; IOFFSET adds to both addresses) --
__device__ __forceinline__ void stage_tile_async(const __bf16* gsrc_lane, unsigned lds_lane) {
#pragma unroll
  for (int j = 0; j < 16; ++j)
    asm volatile("global_load_async_to_lds_b128 %0, %1, off offset:%2"
                 :: "v"(lds_lane), "v"(gsrc_lane), "i"(j * 512) : "memory");
}
__device__ __forceinline__ void wait_async(int n) {
  if (n == 0) asm volatile("s_wait_asynccnt 0x0" ::: "memory");
  else        asm volatile("s_wait_asynccnt 0x10" ::: "memory");
}

// ---- kernel 0: convert weights fp32 -> bf16 once ----------------------------
__global__ __launch_bounds__(256) void cvt_w_kernel(
    const float* __restrict__ Wq, const float* __restrict__ Wk,
    __bf16* __restrict__ Wqbf, __bf16* __restrict__ Wkbf) {
  const int i = blockIdx.x * 256 + threadIdx.x;  // 65536 threads
  Wqbf[i] = (__bf16)Wq[i];
  Wkbf[i] = (__bf16)Wk[i];
}

// ---- kernel 1: Y = bf16( X @ W^T + bias ) -----------------------------------
// W (256x256 bf16 = 128KB) is made LDS-resident via async copies, so the
// inner loop feeds WMMA B-fragments at LDS latency. The fp32 A stream (F_s,
// read exactly once) is double-buffered in registers: chunk kc+1 global loads
// overlap the 16 WMMAs of chunk kc. Each wave produces a 16x256 output tile.
__global__ __launch_bounds__(256) void proj_bf16_kernel(
    const float* __restrict__ X, const __bf16* __restrict__ Wbf,
    const float* __restrict__ bias, __bf16* __restrict__ Y)
{
  __shared__ __bf16 wlds[DK * DK];  // 128 KB: entire W
  const int tid  = threadIdx.x;
  const int lane = tid & 31;
  const int wave = tid >> 5;
  const int h    = lane >> 4;
  const int lr   = lane & 15;
  const int m0   = blockIdx.x * 128 + wave * 16;

  // Stage W into LDS: 256 threads x 16B x 32 rounds = 131072 B, linear copy.
  {
    const __bf16* g = Wbf + (size_t)tid * 8;  // +tid*16 bytes
    unsigned l = (unsigned)(uintptr_t)(&wlds[0]) + (unsigned)tid * 16u;
#pragma unroll
    for (int j = 0; j < 32; ++j)
      asm volatile("global_load_async_to_lds_b128 %0, %1, off offset:%2"
                   :: "v"(l), "v"(g), "i"(j * 4096) : "memory");
    asm volatile("s_wait_asynccnt 0x0" ::: "memory");
  }
  __syncthreads();

  const float* arow = X + (size_t)(m0 + lr) * DK;

  v8f acc[16];
#pragma unroll
  for (int t = 0; t < 16; ++t) acc[t] = zero8();

  v16bf af = loadA_f32(arow, 0, h);
#pragma unroll
  for (int kc = 0; kc < 8; ++kc) {
    v16bf anx = af;
    if (kc < 7) anx = loadA_f32(arow, kc + 1, h);  // overlap with WMMAs below
#pragma unroll
    for (int t = 0; t < 16; t += 2) {
      v16bf b0 = loadB_bf(&wlds[(size_t)(t * 16 + lr) * DK], kc, h);
      v16bf b1 = loadB_bf(&wlds[(size_t)((t + 1) * 16 + lr) * DK], kc, h);
      acc[t] = __builtin_amdgcn_wmma_f32_16x16x32_bf16(
          false, af, false, b0, (short)0, acc[t], false, false);
      acc[t + 1] = __builtin_amdgcn_wmma_f32_16x16x32_bf16(
          false, af, false, b1, (short)0, acc[t + 1], false, false);
    }
    af = anx;
  }

#pragma unroll
  for (int t = 0; t < 16; ++t) {
    const int n  = t * 16 + lr;
    const float bv = bias[n];
#pragma unroll
    for (int i = 0; i < 8; ++i) {
      const int m = m0 + i + h * 8;  // D layout: row = vgpr + 8*(lane>=16)
      Y[(size_t)m * DK + n] = (__bf16)(acc[t][i] + bv);
    }
  }
}

// ---- kernel 2: QK^T/sqrt(d) + mask + softmax over HW ------------------------
// One WG owns a 16-row x 4096-col score tile (contiguous 256 KB of out).
// K tiles stream to LDS with double-buffered async copies (ASYNCcnt), WMMA
// consumes B-fragments from LDS; online max/sum per row via 16-lane
// butterflies; then normalize the tile in place (L2-hot re-read).
__global__ __launch_bounds__(256) void attn_softmax_kernel(
    const __bf16* __restrict__ Qbf, const __bf16* __restrict__ Kbf,
    const int* __restrict__ Ms, float* __restrict__ out)
{
  __shared__ __bf16 kbuf[2][8][16 * DK];  // 2 stages x 8 waves x 8KB = 128KB
  __shared__ float smax[8][16], ssum[8][16], fmx[16], fri[16];

  const int lane = threadIdx.x & 31;
  const int wave = threadIdx.x >> 5;
  const int h    = lane >> 4;
  const int lr   = lane & 15;
  const int b    = blockIdx.y;
  const int mt   = blockIdx.x;

  const __bf16* Qb   = Qbf + ((size_t)b * TN + (size_t)mt * 16) * DK;
  const __bf16* Kb   = Kbf + (size_t)b * HWN * DK;
  const int*    mrow = Ms + (size_t)b * HWN;
  float*        orow = out + ((size_t)b * TN + (size_t)mt * 16) * HWN;

  // Preload the 16x256 Q tile as 8 A-fragments (lives in VGPRs all kernel).
  v16bf a[8];
  {
    const __bf16* qr = Qb + (size_t)lr * DK;
#pragma unroll
    for (int kc = 0; kc < 8; ++kc) a[kc] = loadA_bf(qr, kc, h);
  }

  unsigned ldsw[2];
  ldsw[0] = (unsigned)(uintptr_t)(&kbuf[0][wave][0]) + (unsigned)lane * 16u;
  ldsw[1] = (unsigned)(uintptr_t)(&kbuf[1][wave][0]) + (unsigned)lane * 16u;

  float rm[8], rs[8];
#pragma unroll
  for (int i = 0; i < 8; ++i) { rm[i] = -1e30f; rs[i] = 0.0f; }

  // Prologue: stage first tile.
  stage_tile_async(Kb + (size_t)(wave * 16) * DK + lane * 8, ldsw[0]);

  int stage = 0;
  for (int nt = wave * 16; nt < HWN; nt += 128) {
    if (nt + 128 < HWN) {
      stage_tile_async(Kb + (size_t)(nt + 128) * DK + lane * 8, ldsw[stage ^ 1]);
      wait_async(16);   // current tile done; next tile's 16 copies in flight
    } else {
      wait_async(0);
    }

    const __bf16* krow = &kbuf[stage][wave][lr * DK];  // B column = K row lr
    v8f acc = zero8();
#pragma unroll
    for (int kc = 0; kc < 8; ++kc) {
      v16bf bfrag = loadB_bf(krow, kc, h);
      acc = __builtin_amdgcn_wmma_f32_16x16x32_bf16(
          false, a[kc], false, bfrag, (short)0, acc, false, false);
    }

    const int  n    = nt + lr;
    const bool live = (mrow[n] != 0);
#pragma unroll
    for (int i = 0; i < 8; ++i) {
      float x = live ? acc[i] * 0.0625f : -1e30f;  // 1/sqrt(256); mask
      orow[(size_t)(i + h * 8) * HWN + n] = x;     // raw score
      float cm = x;
#pragma unroll
      for (int s = 1; s < 16; s <<= 1) cm = fmaxf(cm, __shfl_xor(cm, s, 16));
      const float nm = fmaxf(rm[i], cm);
      float p = __expf(x - nm);
#pragma unroll
      for (int s = 1; s < 16; s <<= 1) p += __shfl_xor(p, s, 16);
      rs[i] = rs[i] * __expf(rm[i] - nm) + p;
      rm[i] = nm;
    }
    stage ^= 1;
  }

  // Cross-wave stat combine.
  if (lr == 0) {
#pragma unroll
    for (int i = 0; i < 8; ++i) {
      smax[wave][i + h * 8] = rm[i];
      ssum[wave][i + h * 8] = rs[i];
    }
  }
  __syncthreads();
  if (threadIdx.x < 16) {
    const int row = threadIdx.x;
    float M = -1e30f;
#pragma unroll
    for (int w = 0; w < 8; ++w) M = fmaxf(M, smax[w][row]);
    float S = 0.0f;
#pragma unroll
    for (int w = 0; w < 8; ++w) S += ssum[w][row] * __expf(smax[w][row] - M);
    fmx[row] = M;
    fri[row] = 1.0f / S;
  }
  __threadfence();
  __syncthreads();

  // Normalize the contiguous 16x4096 tile in place (L2-hot re-read).
  for (int base = threadIdx.x * 4; base < 16 * HWN; base += 256 * 4) {
    const int row = base >> 12;
    float4 v = *(const float4*)(orow + base);
    const float M = fmx[row], r = fri[row];
    v.x = __expf(v.x - M) * r;
    v.y = __expf(v.y - M) * r;
    v.z = __expf(v.z - M) * r;
    v.w = __expf(v.w - M) * r;
    *(float4*)(orow + base) = v;
  }
}

extern "C" void kernel_launch(void* const* d_in, const int* in_sizes, int n_in,
                              void* d_out, int out_size, void* d_ws, size_t ws_size,
                              hipStream_t stream) {
  const float* F_a = (const float*)d_in[0];  // [32,256,256]
  const float* F_s = (const float*)d_in[1];  // [32,4096,256]
  const int*   M_s = (const int*)d_in[2];    // [32,1,64,64] -> [32,4096]
  const float* Wq  = (const float*)d_in[3];
  const float* bq  = (const float*)d_in[4];
  const float* Wk  = (const float*)d_in[5];
  const float* bk  = (const float*)d_in[6];
  float* out = (float*)d_out;

  __bf16* Qbf  = (__bf16*)d_ws;                       // 8192*256   bf16 (4 MB)
  __bf16* Kbf  = Qbf + (size_t)BN * TN * DK;          // 131072*256 bf16 (67 MB)
  __bf16* Wqbf = Kbf + (size_t)BN * HWN * DK;         // 256*256 bf16
  __bf16* Wkbf = Wqbf + (size_t)DK * DK;

  dim3 blk(256);
  cvt_w_kernel<<<dim3(DK * DK / 256), blk, 0, stream>>>(Wq, Wk, Wqbf, Wkbf);
  proj_bf16_kernel<<<dim3((BN * TN) / 128), blk, 0, stream>>>(F_a, Wqbf, bq, Qbf);
  proj_bf16_kernel<<<dim3((BN * HWN) / 128), blk, 0, stream>>>(F_s, Wkbf, bk, Kbf);
  attn_softmax_kernel<<<dim3(TN / 16, BN), blk, 0, stream>>>(Qbf, Kbf, M_s, out);
}
